// NodeModel_54451595379231
// MI455X (gfx1250) — compile-verified, matrix operand order checked by reading, not run on hardware
//
#include <hip/hip_runtime.h>
#include <hip/hip_bf16.h>

#define HIDDEN 128
#define NNODES 50000
#define NEDGES 625000
#define NTILES (NNODES / 16)   // 3125
#define WAVES_PER_BLOCK 4

typedef __attribute__((ext_vector_type(16))) __bf16 v16bf;
typedef __attribute__((ext_vector_type(8)))  float  v8f;

// f32 -> bf16 round-to-nearest-even
__device__ __forceinline__ unsigned short f2bf(float f) {
  unsigned u = __float_as_uint(f);
  unsigned r = u + 0x7FFFu + ((u >> 16) & 1u);
  return (unsigned short)(r >> 16);
}
__device__ __forceinline__ unsigned pack2(float lo, float hi) {
  return (unsigned)f2bf(lo) | ((unsigned)f2bf(hi) << 16);
}

union Frag {
  v16bf   v;
  unsigned u[8];
  uint4    q[2];
};

__device__ __forceinline__ v8f splat8(float s) {
  v8f a;
#pragma unroll
  for (int i = 0; i < 8; ++i) a[i] = s;
  return a;
}

// ---------------------------------------------------------------- kernel 0
__global__ void zero_ws_kernel(float4* __restrict__ p, int n4) {
  int i = blockIdx.x * blockDim.x + threadIdx.x;
  if (i < n4) p[i] = make_float4(0.f, 0.f, 0.f, 0.f);
}

// ---------------------------------------------------------------- kernel 1
// scatter-add edge_attr rows into esum[row] via hardware f32 atomics.
// one thread per 4 features of one edge (E*32 threads total).
__global__ void scatter_kernel(const float* __restrict__ edge_attr,
                               const int* __restrict__ row_idx,
                               float* __restrict__ esum) {
  int gid = blockIdx.x * blockDim.x + threadIdx.x;
  int e = gid >> 5;
  if (e >= NEDGES) return;
  int c = (gid & 31) << 2;
  float4 v = *(const float4*)(edge_attr + (size_t)e * HIDDEN + c);
  int r = row_idx[e];
  float* dst = esum + (size_t)r * HIDDEN + c;
  unsafeAtomicAdd(dst + 0, v.x);
  unsafeAtomicAdd(dst + 1, v.y);
  unsafeAtomicAdd(dst + 2, v.z);
  unsafeAtomicAdd(dst + 3, v.w);
}

// ---------------------------------------------------------------- kernel 2
// fused: out = relu(concat(x, esum) @ W1 + b1) @ W2 + b2 + x
// per-wave 16-node tile, WMMA bf16 16x16x32, weights staged in LDS (bf16, transposed)
__global__ __launch_bounds__(128) void fused_mlp_kernel(
    const float* __restrict__ x, const float* __restrict__ esum,
    const float* __restrict__ W1, const float* __restrict__ b1,
    const float* __restrict__ W2, const float* __restrict__ b2,
    float* __restrict__ out) {
  extern __shared__ char smem[];
  unsigned short* w1t = (unsigned short*)smem;            // [128][256] bf16: w1t[n][k] = W1[k][n]
  unsigned short* w2t = w1t + 128 * 256;                  // [128][128] bf16: w2t[n][k] = W2[k][n]
  float* b1s = (float*)(w2t + 128 * 128);                 // [128]
  float* b2s = b1s + 128;                                 // [128]
  unsigned short* hid = (unsigned short*)(b2s + 128);     // [4][16][128] bf16 hidden tiles

  const int tid = threadIdx.x;

  // cooperative weight staging (transpose + f32->bf16)
  for (int i = tid; i < 256 * 128; i += 128) {
    int k = i >> 7, n = i & 127;
    w1t[n * 256 + k] = f2bf(W1[i]);
  }
  for (int i = tid; i < 128 * 128; i += 128) {
    int k = i >> 7, n = i & 127;
    w2t[n * 128 + k] = f2bf(W2[i]);
  }
  b1s[tid] = b1[tid];   // blockDim == 128
  b2s[tid] = b2[tid];
  __syncthreads();

  const int lane = tid & 31;
  const int wave = tid >> 5;
  const int tile = blockIdx.x * WAVES_PER_BLOCK + wave;
  if (tile >= NTILES) return;

  const int row0 = tile * 16;
  const int mrow = lane & 15;   // A row / D column within 16-tile
  const int g    = lane >> 4;   // K-half selector per ISA 16-bit A layout
  unsigned short* hw = hid + wave * (16 * HIDDEN);

  // ---- stage 1: preload all 8 A fragments of concat(x, esum) (K = 256)
  Frag afr[8];
#pragma unroll
  for (int ks = 0; ks < 8; ++ks) {
    const float* src = (ks < 4) ? x : esum;
    const int col0 = (ks < 4) ? ks * 32 : ks * 32 - 128;
    const float* pa = src + (size_t)(row0 + mrow) * HIDDEN + col0 + g * 8;
    float4 a0 = *(const float4*)(pa + 0);    // K offs g*8 + 0..3
    float4 a1 = *(const float4*)(pa + 4);    // K offs g*8 + 4..7
    float4 a2 = *(const float4*)(pa + 16);   // K offs 16 + g*8 + 0..3
    float4 a3 = *(const float4*)(pa + 20);   // K offs 16 + g*8 + 4..7
    afr[ks].u[0] = pack2(a0.x, a0.y);
    afr[ks].u[1] = pack2(a0.z, a0.w);
    afr[ks].u[2] = pack2(a1.x, a1.y);
    afr[ks].u[3] = pack2(a1.z, a1.w);
    afr[ks].u[4] = pack2(a2.x, a2.y);
    afr[ks].u[5] = pack2(a2.z, a2.w);
    afr[ks].u[6] = pack2(a3.x, a3.y);
    afr[ks].u[7] = pack2(a3.z, a3.w);
  }

  // layer 1: hidden[16][128] = relu(A @ W1 + b1), kept as bf16 in LDS
#pragma unroll
  for (int nt = 0; nt < 8; ++nt) {
    const int col = nt * 16 + mrow;
    v8f acc = splat8(b1s[col]);
#pragma unroll
    for (int ks = 0; ks < 8; ++ks) {
      Frag bfr;
      const unsigned short* wb = w1t + col * 256 + ks * 32 + g * 8;
      bfr.q[0] = *(const uint4*)(wb);        // K offs g*8 + 0..7
      bfr.q[1] = *(const uint4*)(wb + 16);   // K offs 16 + g*8 + 0..7
      acc = __builtin_amdgcn_wmma_f32_16x16x32_bf16(
          false, afr[ks].v, false, bfr.v, (short)0, acc, false, false);
    }
#pragma unroll
    for (int r = 0; r < 8; ++r) {
      float h = acc[r] > 0.f ? acc[r] : 0.f;
      hw[(r + g * 8) * HIDDEN + col] = f2bf(h);   // D-layout -> row-major LDS
    }
  }

  // layer 2: out = hidden @ W2 + b2 + x   (K = 128)
#pragma unroll
  for (int nt = 0; nt < 8; ++nt) {
    const int col = nt * 16 + mrow;
    v8f acc = splat8(b2s[col]);
#pragma unroll
    for (int ks = 0; ks < 4; ++ks) {
      Frag af, bf_;
      const unsigned short* pa = hw + mrow * HIDDEN + ks * 32 + g * 8;
      af.q[0] = *(const uint4*)(pa);
      af.q[1] = *(const uint4*)(pa + 16);
      const unsigned short* pb = w2t + col * 128 + ks * 32 + g * 8;
      bf_.q[0] = *(const uint4*)(pb);
      bf_.q[1] = *(const uint4*)(pb + 16);
      acc = __builtin_amdgcn_wmma_f32_16x16x32_bf16(
          false, af.v, false, bf_.v, (short)0, acc, false, false);
    }
#pragma unroll
    for (int r = 0; r < 8; ++r) {
      size_t idx = (size_t)(row0 + r + g * 8) * HIDDEN + col;
      out[idx] = acc[r] + x[idx];   // residual
    }
  }
}

// ---------------------------------------------------------------- host
extern "C" void kernel_launch(void* const* d_in, const int* in_sizes, int n_in,
                              void* d_out, int out_size, void* d_ws, size_t ws_size,
                              hipStream_t stream) {
  const float* x   = (const float*)d_in[0];
  const int*   ei  = (const int*)d_in[1];    // edge_index[0][e] = source row, first E entries
  const float* ea  = (const float*)d_in[2];
  // d_in[3] = u, d_in[4] = batch : unused by the reference output
  const float* W1  = (const float*)d_in[5];
  const float* b1  = (const float*)d_in[6];
  const float* W2  = (const float*)d_in[7];
  const float* b2  = (const float*)d_in[8];
  float* out  = (float*)d_out;
  float* esum = (float*)d_ws;   // N*HIDDEN f32 = 25.6 MB scratch

  // 0) zero the segment-sum accumulator (ws is poisoned; must zero every call)
  int n4 = NNODES * HIDDEN / 4;
  zero_ws_kernel<<<(n4 + 255) / 256, 256, 0, stream>>>((float4*)esum, n4);

  // 1) scatter-add edges (hardware global_atomic_add_f32)
  int sthreads = NEDGES * 32;
  scatter_kernel<<<(sthreads + 255) / 256, 256, 0, stream>>>(ea, ei, esum);

  // 2) fused 2-layer MLP with WMMA bf16 + residual
  size_t smem = (size_t)128 * 256 * 2   // w1t
              + (size_t)128 * 128 * 2   // w2t
              + 2 * 128 * 4             // biases
              + (size_t)WAVES_PER_BLOCK * 16 * HIDDEN * 2;  // hidden tiles
  int blocks = (NTILES + WAVES_PER_BLOCK - 1) / WAVES_PER_BLOCK;
  fused_mlp_kernel<<<blocks, 128, smem, stream>>>(x, esum, W1, b1, W2, b2, out);
}
